// SelfAttention_33088428048737
// MI455X (gfx1250) — matrix operand order repaired and checked by reading, not compiled, not run on hardware
//
#include <hip/hip_runtime.h>
#include <hip/hip_bf16.h>
#include <math.h>

typedef __bf16 bf16_t;
typedef __attribute__((ext_vector_type(16))) __bf16 v16bf;
typedef __attribute__((ext_vector_type(8)))  __bf16 v8bf;
typedef __attribute__((ext_vector_type(8)))  float  v8f;

#define C_DIM 256
#define D_QK  32
#define N_B   4
#define N_S   4096   // 64*64

static __device__ __forceinline__ int lane_id() { return (int)(threadIdx.x & 31); }

// ---- WMMA fragment loaders (wave32, 16x16x32 bf16) --------------------------
// A matrix 16x32 (MxK), row-major source with leading dim ld.
// Lane (h*16+m) holds row m: K in [h*16, h*16+16) contiguous.
static __device__ __forceinline__ v16bf load_A16x32(const bf16_t* p, int ld) {
  int lane = lane_id();
  int m = lane & 15;
  int h = lane >> 4;
  const bf16_t* row = p + (long)m * ld;
  return *(const v16bf*)(row + h * 16);
}

// B matrix 32x16 (KxN), with element B[k][col] = p[col*ld + k]  (k contiguous).
// Lane (hk*16+col) holds column col, K in [hk*16, hk*16+16) contiguous.
static __device__ __forceinline__ v16bf load_B32x16(const bf16_t* p, int ld) {
  int lane = lane_id();
  int col = lane & 15;
  int hk = lane >> 4;
  return *(const v16bf*)(p + (long)col * ld + hk * 16);
}

static __device__ __forceinline__ v8f wmma_bf16(v16bf a, v16bf b, v8f c) {
  return __builtin_amdgcn_wmma_f32_16x16x32_bf16(false, a, false, b, (short)0, c,
                                                 false, false);
}

// Row reductions: C/D layout keeps each 16-wide row inside one 16-lane half.
static __device__ __forceinline__ float red16_max(float v) {
#pragma unroll
  for (int m = 1; m < 16; m <<= 1) v = fmaxf(v, __shfl_xor(v, m, 32));
  return v;
}
static __device__ __forceinline__ float red16_sum(float v) {
#pragma unroll
  for (int m = 1; m < 16; m <<= 1) v += __shfl_xor(v, m, 32);
  return v;
}

// ---- Kernel 0: convert x to bf16 transposed [B][N][C]; convert weights -----
__global__ void prep_kernel(const float* __restrict__ x,
                            const float* __restrict__ Wq,
                            const float* __restrict__ Wk,
                            const float* __restrict__ Wv,
                            bf16_t* __restrict__ xt,   // [B][N][C]
                            bf16_t* __restrict__ wqb,  // [32][256]
                            bf16_t* __restrict__ wkb,  // [32][256]
                            bf16_t* __restrict__ wvb)  // [256][256]
{
  const long NXT  = (long)N_B * N_S * C_DIM;
  const long NWQK = (long)D_QK * C_DIM;
  const long NWV  = (long)C_DIM * C_DIM;
  long i = (long)blockIdx.x * blockDim.x + threadIdx.x;
  if (i < NXT) {
    int c = (int)(i & (C_DIM - 1));
    long bn = i >> 8;
    int n = (int)(bn & (N_S - 1));
    int b = (int)(bn >> 12);
    xt[i] = (bf16_t)x[(long)(b * C_DIM + c) * N_S + n];
    return;
  }
  i -= NXT;
  if (i < NWQK) { wqb[i] = (bf16_t)Wq[i]; return; }
  i -= NWQK;
  if (i < NWQK) { wkb[i] = (bf16_t)Wk[i]; return; }
  i -= NWQK;
  if (i < NWV)  { wvb[i] = (bf16_t)Wv[i]; return; }
}

// ---- Kernel 1: Q/K projection.  Q[b][n][d], K[b][n][d] (bf16) --------------
__global__ void __launch_bounds__(128) qk_proj_kernel(
    const bf16_t* __restrict__ xt, const bf16_t* __restrict__ wqb,
    const bf16_t* __restrict__ wkb, const float* __restrict__ bq,
    const float* __restrict__ bk, bf16_t* __restrict__ Q,
    bf16_t* __restrict__ K)
{
  int wave = (int)((blockIdx.x * blockDim.x + threadIdx.x) >> 5);
  int lane = lane_id();
  int b = wave >> 8;
  int n0 = (wave & 255) << 4;
  const bf16_t* xb = xt + ((long)b * N_S + n0) * C_DIM;
  v8f q0 = {}, q1 = {}, k0 = {}, k1 = {};
#pragma unroll
  for (int c0 = 0; c0 < C_DIM; c0 += 32) {
    v16bf a   = load_A16x32(xb + c0, C_DIM);
    v16bf bq0 = load_B32x16(wqb + c0, C_DIM);
    v16bf bq1 = load_B32x16(wqb + 16 * C_DIM + c0, C_DIM);
    v16bf bk0 = load_B32x16(wkb + c0, C_DIM);
    v16bf bk1 = load_B32x16(wkb + 16 * C_DIM + c0, C_DIM);
    q0 = wmma_bf16(a, bq0, q0);
    q1 = wmma_bf16(a, bq1, q1);
    k0 = wmma_bf16(a, bk0, k0);
    k1 = wmma_bf16(a, bk1, k1);
  }
  int col = lane & 15, h = lane >> 4;
  float biasq0 = bq[col], biasq1 = bq[16 + col];
  float biask0 = bk[col], biask1 = bk[16 + col];
  bf16_t* Qo = Q + ((long)b * N_S + n0) * D_QK;
  bf16_t* Ko = K + ((long)b * N_S + n0) * D_QK;
#pragma unroll
  for (int r = 0; r < 8; ++r) {
    int m = r + 8 * h;
    Qo[m * D_QK + col]      = (bf16_t)(q0[r] + biasq0);
    Qo[m * D_QK + 16 + col] = (bf16_t)(q1[r] + biasq1);
    Ko[m * D_QK + col]      = (bf16_t)(k0[r] + biask0);
    Ko[m * D_QK + 16 + col] = (bf16_t)(k1[r] + biask1);
  }
}

// ---- Kernel 2: V projection, channel-major V[b][e][n] (bf16) ---------------
__global__ void __launch_bounds__(128) v_proj_kernel(
    const bf16_t* __restrict__ xt, const bf16_t* __restrict__ wvb,
    const float* __restrict__ bv, bf16_t* __restrict__ V)
{
  int wave = (int)((blockIdx.x * blockDim.x + threadIdx.x) >> 5);
  int lane = lane_id();
  int nt = wave & 255;
  int et = (wave >> 8) & 15;
  int b  = wave >> 12;
  int n0 = nt << 4, e0 = et << 4;
  const bf16_t* xb = xt + ((long)b * N_S + n0) * C_DIM;
  v8f acc = {};
#pragma unroll
  for (int c0 = 0; c0 < C_DIM; c0 += 32) {
    v16bf a  = load_A16x32(wvb + (long)e0 * C_DIM + c0, C_DIM);
    v16bf bm = load_B32x16(xb + c0, C_DIM);
    acc = wmma_bf16(a, bm, acc);
  }
  int col = lane & 15, h = lane >> 4;
  bf16_t* Vo = V + (long)b * C_DIM * N_S;
#pragma unroll
  for (int r = 0; r < 8; ++r) {
    int e = e0 + r + 8 * h;
    Vo[(long)e * N_S + n0 + col] = (bf16_t)(acc[r] + bv[e]);
  }
}

// ---- Kernel 3: flash attention with async-LDS double buffering -------------
// 4 waves/block share one batch; K/V chunks staged once per block via
// global_load_async_to_lds_b128 (ASYNCcnt), double-buffered.
__global__ void __launch_bounds__(128, 1) attn_kernel(
    const bf16_t* __restrict__ Q, const bf16_t* __restrict__ K,
    const bf16_t* __restrict__ V, const float* __restrict__ x,
    const float* __restrict__ gamma, float* __restrict__ out)
{
  __shared__ __align__(32) bf16_t Vbuf[2][C_DIM * 32];  // [e][k] rows of 64B
  __shared__ __align__(32) bf16_t Kbuf[2][32 * 32];     // [key][d]
  __shared__ __align__(32) bf16_t pstage[4][16 * 32];   // P staging, per wave
  __shared__ __align__(32) float  ostage[4][16 * 16];   // epilogue transpose

  int wib  = (int)(threadIdx.x >> 5);
  int lane = lane_id();
  int wave = (int)blockIdx.x * 4 + wib;
  int b  = wave >> 8;
  int n0 = (wave & 255) << 4;
  int col = lane & 15, h = lane >> 4;

  const bf16_t* Qg = Q + (long)b * N_S * D_QK;
  const char*   Kg = (const char*)(K + (long)b * N_S * D_QK);
  const char*   Vg = (const char*)(V + (long)b * C_DIM * N_S);

  v16bf qa = load_A16x32(Qg + (long)n0 * D_QK, D_QK);

  v8f o[16];
#pragma unroll
  for (int t = 0; t < 16; ++t) o[t] = (v8f){};
  float mrow[8], lrow[8];
#pragma unroll
  for (int r = 0; r < 8; ++r) { mrow[r] = -3.0e38f; lrow[r] = 0.0f; }

  bf16_t* P = &pstage[wib][0];

  // Per-block async stage of one 32-key chunk: 8 V issues + 1 K issue / wave.
  auto issue_chunk = [&](int kc, int bufi) {
    unsigned vl = (unsigned)(uintptr_t)(&Vbuf[bufi][0]);
    unsigned kl = (unsigned)(uintptr_t)(&Kbuf[bufi][0]);
#pragma unroll
    for (int j = 0; j < 8; ++j) {
      int t = wib * 256 + j * 32 + lane;   // [0,1024): 256 e-rows x 4 quads
      int e = t >> 2, q = t & 3;
      unsigned long long ga =
          (unsigned long long)(uintptr_t)(Vg + ((long)e * N_S + kc) * 2 + q * 16);
      unsigned la = vl + (unsigned)(e * 64 + q * 16);
      asm volatile("global_load_async_to_lds_b128 %0, %1, off"
                   :: "v"(la), "v"(ga) : "memory");
    }
    {
      int u = wib * 32 + lane;             // [0,128): 2KB K chunk
      unsigned long long ga =
          (unsigned long long)(uintptr_t)(Kg + (long)kc * 64 + u * 16);
      unsigned la = kl + (unsigned)(u * 16);
      asm volatile("global_load_async_to_lds_b128 %0, %1, off"
                   :: "v"(la), "v"(ga) : "memory");
    }
  };

  issue_chunk(0, 0);
  const int NCH = N_S / 32;  // 128 chunks
  for (int ic = 0; ic < NCH; ++ic) {
    int cur = ic & 1;
    if (ic + 1 < NCH) {
      issue_chunk((ic + 1) * 32, cur ^ 1);
      asm volatile("s_wait_asynccnt 0x9" ::: "memory");  // chunk ic landed
    } else {
      asm volatile("s_wait_asynccnt 0x0" ::: "memory");
    }
    __syncthreads();  // all waves' portions of chunk ic visible

    const bf16_t* Kc = &Kbuf[cur][0];
    const bf16_t* Vc = &Vbuf[cur][0];

    // S = Q * K^T for 32 keys (K-dim = d = 32 exactly)
    v16bf kb0 = load_B32x16(Kc, 32);
    v16bf kb1 = load_B32x16(Kc + 16 * 32, 32);
    v8f z = {};
    v8f s0 = wmma_bf16(qa, kb0, z);
    v8f s1 = wmma_bf16(qa, kb1, z);

    // online softmax (row m = r + 8*h lives in one 16-lane half)
#pragma unroll
    for (int r = 0; r < 8; ++r) {
      float mx = red16_max(fmaxf(s0[r], s1[r]));
      float mnew = fmaxf(mrow[r], mx);
      float corr = __expf(mrow[r] - mnew);
      float p0 = __expf(s0[r] - mnew);
      float p1 = __expf(s1[r] - mnew);
      lrow[r] = lrow[r] * corr + red16_sum(p0 + p1);
      mrow[r] = mnew;
      int m = r + 8 * h;
      P[m * 32 + col]      = (bf16_t)p0;
      P[m * 32 + 16 + col] = (bf16_t)p1;
#pragma unroll
      for (int t = 0; t < 16; ++t) o[t][r] *= corr;
    }

    asm volatile("s_wait_dscnt 0x0" ::: "memory");  // P stores visible

    // O += P * V : A = P (16x32), B = V chunk [32 keys x 16 channels]
    v16bf pa = load_A16x32(P, 32);
#pragma unroll
    for (int t = 0; t < 16; ++t) {
      v16bf vbf = load_B32x16(Vc + t * 16 * 32, 32);
      o[t] = wmma_bf16(pa, vbf, o[t]);
    }
    __syncthreads();  // done reading buf[cur] before it is refilled
  }

  // epilogue: normalize, transpose tiles through LDS, coalesced y = g*O + x
  float g = gamma[0];
  float inv[8];
#pragma unroll
  for (int r = 0; r < 8; ++r) inv[r] = 1.0f / lrow[r];
  float* ptile = &ostage[wib][0];
  int rrow = lane >> 1;   // channel-within-tile 0..15
  int seg  = lane & 1;    // 8-float segment 0..1
#pragma unroll
  for (int t = 0; t < 16; ++t) {
#pragma unroll
    for (int r = 0; r < 8; ++r)
      ptile[(r + 8 * h) * 16 + col] = o[t][r] * inv[r];  // [m][c'] scatter
    // LDS ops are in-order within a wave: the loads below see the stores.
    int c = t * 16 + rrow;
    long base = (long)(b * C_DIM + c) * N_S + n0 + seg * 8;
    v8f ov = *(const v8f*)(ptile + rrow * 16 + seg * 8);
    v8f xv = *(const v8f*)(x + base);
    v8f yv;
#pragma unroll
    for (int i = 0; i < 8; ++i) yv[i] = g * ov[i] + xv[i];
    *(v8f*)(out + base) = yv;
  }
}

// ---- host-side launcher -----------------------------------------------------
extern "C" void kernel_launch(void* const* d_in, const int* in_sizes, int n_in,
                              void* d_out, int out_size, void* d_ws, size_t ws_size,
                              hipStream_t stream) {
  (void)in_sizes; (void)n_in; (void)out_size; (void)ws_size;
  const float* x     = (const float*)d_in[0];
  const float* Wq    = (const float*)d_in[1];
  const float* bq    = (const float*)d_in[2];
  const float* Wk    = (const float*)d_in[3];
  const float* bk    = (const float*)d_in[4];
  const float* Wv    = (const float*)d_in[5];
  const float* bv    = (const float*)d_in[6];
  const float* gamma = (const float*)d_in[7];
  float* out = (float*)d_out;

  char* ws = (char*)d_ws;                       // offsets all 256B-aligned
  bf16_t* xt  = (bf16_t*)(ws + 0);              //  8,388,608  [B][N][C]
  bf16_t* wqb = (bf16_t*)(ws + 8388608);        //     16,384  [32][256]
  bf16_t* wkb = (bf16_t*)(ws + 8404992);        //     16,384  [32][256]
  bf16_t* wvb = (bf16_t*)(ws + 8421376);        //    131,072  [256][256]
  bf16_t* Qb  = (bf16_t*)(ws + 8552448);        //  1,048,576  [B][N][32]
  bf16_t* Kb  = (bf16_t*)(ws + 9601024);        //  1,048,576  [B][N][32]
  bf16_t* Vb  = (bf16_t*)(ws + 10649600);       //  8,388,608  [B][C][N]

  long total_prep = (long)N_B * N_S * C_DIM + 2L * D_QK * C_DIM + (long)C_DIM * C_DIM;
  int pb = (int)((total_prep + 255) / 256);
  prep_kernel<<<pb, 256, 0, stream>>>(x, Wq, Wk, Wv, xt, wqb, wkb, wvb);

  qk_proj_kernel<<<256, 128, 0, stream>>>(xt, wqb, wkb, bq, bk, Qb, Kb);
  v_proj_kernel<<<4096, 128, 0, stream>>>(xt, wvb, bv, Vb);
  attn_kernel<<<256, 128, 0, stream>>>(Qb, Kb, Vb, x, gamma, out);
}
